// MoE_lora_new_88424786690149
// MI455X (gfx1250) — matrix operand order repaired and checked by reading, not compiled, not run on hardware
//
#include <hip/hip_runtime.h>

typedef __attribute__((ext_vector_type(16))) _Float16 v16h;
typedef __attribute__((ext_vector_type(8)))  _Float16 v8h;
typedef __attribute__((ext_vector_type(8)))  float    v8f;
typedef __attribute__((ext_vector_type(4)))  unsigned int v4u;
typedef __attribute__((ext_vector_type(4)))  int v4i;
typedef __attribute__((ext_vector_type(8)))  int v8i;

#define B_    32
#define C1_   64
#define HW_   64
#define C2_   128
#define KDIM  576      // C1*3*3
#define E_    4
#define NTILE 128
#define KTILE 32

#if defined(__has_builtin)
#if __has_builtin(__builtin_amdgcn_tensor_load_to_lds) && \
    __has_builtin(__builtin_amdgcn_s_wait_tensorcnt)
#define USE_TDM 1
#endif
#endif
#ifndef USE_TDM
#define USE_TDM 0
#endif

// ---------------------------------------------------------------------------
// 1) Gating: channel means -> logits -> top-2 softmax -> dense gates [B][E]
// ---------------------------------------------------------------------------
__global__ void moe_gate_kernel(const float* __restrict__ x,
                                const float* __restrict__ w_gate,
                                float* __restrict__ gates) {
  const int b = blockIdx.x;
  const int t = threadIdx.x;
  __shared__ float cs[C1_];
  if (t < C1_) cs[t] = 0.f;
  __syncthreads();
  const int c    = t & 63;
  const int part = t >> 6;                       // 4 partial sums per channel
  const float* p = x + ((size_t)(b * C1_ + c)) * 4096 + (size_t)part * 1024;
  float s = 0.f;
  for (int i = 0; i < 1024; ++i) s += p[i];
  atomicAdd(&cs[c], s);                          // ds_add_f32
  __syncthreads();
  if (t == 0) {
    float logits[E_];
    for (int e = 0; e < E_; ++e) {
      float L = 0.f;
      for (int cc = 0; cc < C1_; ++cc)
        L += (cs[cc] * (1.f / 4096.f)) * w_gate[cc * E_ + e];
      logits[e] = L;
    }
    int i0 = 0;
    for (int e = 1; e < E_; ++e) if (logits[e] > logits[i0]) i0 = e;
    int i1 = -1;
    for (int e = 0; e < E_; ++e) {
      if (e == i0) continue;
      if (i1 < 0 || logits[e] > logits[i1]) i1 = e;
    }
    const float z1  = expf(logits[i1] - logits[i0]);
    const float inv = 1.f / (1.f + z1);
    for (int e = 0; e < E_; ++e) gates[b * E_ + e] = 0.f;
    gates[b * E_ + i0] = inv;
    gates[b * E_ + i1] = z1 * inv;
  }
}

// ---------------------------------------------------------------------------
// 2) fp32 -> f16 conversion of x and all conv weights (already [e][m][k])
// ---------------------------------------------------------------------------
__global__ void moe_convert_kernel(const float* __restrict__ x,
                                   const float* __restrict__ expert_w,
                                   const float* __restrict__ shared_w,
                                   _Float16* __restrict__ xh,
                                   _Float16* __restrict__ wh) {
  const size_t NX  = (size_t)B_ * C1_ * HW_ * HW_;   // 8388608
  const size_t NEW = (size_t)E_ * C2_ * KDIM;        // 294912
  const size_t NW  = NEW + (size_t)C2_ * KDIM;       // 368640
  const size_t stride = (size_t)gridDim.x * blockDim.x;
  for (size_t i = (size_t)blockIdx.x * blockDim.x + threadIdx.x;
       i < NX + NW; i += stride) {
    if (i < NX) {
      xh[i] = (_Float16)x[i];
    } else {
      const size_t w = i - NX;
      const float  v = (w < NEW) ? expert_w[w] : shared_w[w - NEW];
      wh[w] = (_Float16)v;
    }
  }
}

// ---------------------------------------------------------------------------
// TDM: DMA one 128x32-half weight tile into LDS, padding each 32-half row
// (16 DWORDs) with 4 DWORDs to produce the bank-conflict-free [128][40]
// layout.  D# per CDNA5 ISA ch.8: group0 = {count, lds_addr, global_addr,
// type=2}; group1 = {data_size=2B, pad_enable, pad_interval=16dw,
// pad_amount=4dw, tensor_dim0=576, tensor_dim1=128, tile_dim0=32,
// tile_dim1=128, tensor_dim0_stride=576}.
// ---------------------------------------------------------------------------
#if USE_TDM
__device__ __forceinline__ void tdm_load_weight_tile(unsigned lds_off,
                                                     const _Float16* gsrc) {
  const unsigned long long ga = (unsigned long long)(uintptr_t)gsrc;
  const v4u g0 = { 1u,                                    // count=1 (valid D#)
                   lds_off,                               // lds_addr [63:32]
                   (unsigned)(ga & 0xffffffffu),          // global_addr lo
                   (unsigned)((ga >> 32) & 0x01ffffffu)   // global_addr[56:32]
                       | (2u << 30) };                    // type = 2 (image)
  const v8i g1 = {
      (int)((1u << 16) |            // data_size = 1 -> 2 bytes
            (1u << 20) |            // pad_enable
            (3u << 22) |            // pad_interval code 3 -> 16 DWORDs (64B row)
            (3u << 25)),            // pad_amount  code 3 -> 4 DWORDs (8 halfs)
      (int)((KDIM & 0xffffu) << 16),        // tensor_dim0[15:0]=576 @bits[63:48]
      (int)((C2_ & 0xffffu) << 16),         // dim0 hi=0 | tensor_dim1[15:0]=128
      (int)(KTILE << 16),                   // dim1 hi=0 | tile_dim0=32
      (int)NTILE,                           // tile_dim1=128 | tile_dim2=0
      (int)KDIM,                            // tensor_dim0_stride lo = 576
      0, 0 };
  const v4i gz = {0, 0, 0, 0};
#if __clang_major__ >= 23
  const v8i gz8 = {0, 0, 0, 0, 0, 0, 0, 0};
  __builtin_amdgcn_tensor_load_to_lds(g0, g1, gz, gz, gz8, 0);
#else
  __builtin_amdgcn_tensor_load_to_lds(g0, g1, gz, gz, 0);
#endif
}
#endif

// ---------------------------------------------------------------------------
// 3) Implicit-GEMM conv + bias + LayerNorm + gated combine (WMMA f16->f32)
//    block = (image b, 128-position N tile); 8 waves each own 16 M rows.
// ---------------------------------------------------------------------------
__global__ __launch_bounds__(256, 1)
void moe_conv_wmma_kernel(const _Float16* __restrict__ xh,
                          const _Float16* __restrict__ wh,
                          const float* __restrict__ gates,
                          const float* __restrict__ expert_b,
                          const float* __restrict__ eln_w,
                          const float* __restrict__ eln_b,
                          const float* __restrict__ shared_b,
                          const float* __restrict__ sln_w,
                          const float* __restrict__ sln_b,
                          float* __restrict__ out) {
  __shared__ _Float16 As[C2_][40];      // 128 x 32 weights, padded rows (16B aligned)
  __shared__ _Float16 Bs[KTILE][136];   // 32 x 128 im2col,  padded rows (16B aligned)
  __shared__ float csum[NTILE];
  __shared__ float csq[NTILE];

  const int t     = threadIdx.x;
  const int wave  = t >> 5;
  const int lane  = t & 31;
  const int lm    = lane & 15;
  const int lhalf = lane >> 4;
  const int b     = blockIdx.x >> 3;
  const int n0    = (blockIdx.x & 7) * NTILE;

#if USE_TDM
  const unsigned as_lds_off = (unsigned)(uintptr_t)(&As[0][0]);
#endif

  const v8f vzero = {0.f, 0.f, 0.f, 0.f, 0.f, 0.f, 0.f, 0.f};
  v8f oacc[8];
#pragma unroll
  for (int j = 0; j < 8; ++j) oacc[j] = vzero;

#if !USE_TDM
  // A-tile cooperative load mapping (fallback path)
  const int a_row = t >> 1;
  const int a_seg = (t & 1) * 16;
#endif
  // B-tile gather mapping: thread -> (k row, 16-column span)
  const int b_k  = t >> 3;
  const int b_nb = (t & 7) * 16;

#pragma unroll 1
  for (int e = 0; e < E_ + 1; ++e) {
    float gv;
    const float *bias, *gam, *bet;
    const _Float16* wbase;
    if (e < E_) {
      gv    = gates[b * E_ + e];
      bias  = expert_b + e * C2_;
      gam   = eln_w + e * C2_;
      bet   = eln_b + e * C2_;
      wbase = wh + (size_t)e * C2_ * KDIM;
    } else {
      gv   = 1.f;                       // shared expert always active
      bias = shared_b; gam = sln_w; bet = sln_b;
      wbase = wh + (size_t)E_ * C2_ * KDIM;
    }
    // force scalar: uniform branch keeps EXEC all-ones around WMMA
    const float g =
        __int_as_float(__builtin_amdgcn_readfirstlane(__float_as_int(gv)));
    if (g == 0.f) continue;

    v8f acc[8];
#pragma unroll
    for (int j = 0; j < 8; ++j) acc[j] = vzero;

#pragma unroll 1
    for (int kt = 0; kt < KDIM / KTILE; ++kt) {
      const int k0 = kt * KTILE;
      __syncthreads();
#if USE_TDM
      // ---- weight tile via Tensor Data Mover (one wave issues the DMA) ----
      if (t < 32) tdm_load_weight_tile(as_lds_off, wbase + k0);
#else
      // ---- stage weight tile 128x32 with b128 loads ----
      {
        const _Float16* src = wbase + (size_t)a_row * KDIM + k0 + a_seg;
        *(v8h*)&As[a_row][a_seg]     = *(const v8h*)(src);
        *(v8h*)&As[a_row][a_seg + 8] = *(const v8h*)(src + 8);
        if (kt + 1 < KDIM / KTILE) __builtin_prefetch(src + KTILE, 0, 1);
      }
#endif
      // ---- gather im2col tile 32x128 from f16 x (overlaps the DMA) ----
      {
        const int kk = k0 + b_k;
        const int ci = kk / 9;
        const int r9 = kk - ci * 9;
        const int kh = r9 / 3;
        const int kw = r9 - kh * 3;
        const _Float16* xbase = xh + ((size_t)(b * C1_ + ci)) * 4096;
#pragma unroll
        for (int i = 0; i < 16; ++i) {
          const int n  = n0 + b_nb + i;
          const int oh = n >> 5;
          const int ow = n & 31;
          const int ih = oh * 2 - 1 + kh;
          const int iw = ow * 2 - 1 + kw;
          _Float16 v = (_Float16)0.f;
          if ((unsigned)ih < 64u && (unsigned)iw < 64u)
            v = xbase[ih * 64 + iw];
          Bs[b_k][b_nb + i] = v;
        }
      }
#if USE_TDM
      __builtin_amdgcn_s_wait_tensorcnt(0);   // DMA landed in LDS
#endif
      __syncthreads();
      // ---- fragments per documented 16-bit A(16x32)/B(32x16) layouts ----
      const _Float16* arow = &As[wave * 16 + lm][0];
      const v8h alo = *(const v8h*)(arow + lhalf * 8);
      const v8h ahi = *(const v8h*)(arow + 16 + lhalf * 8);
      const v16h a = __builtin_shufflevector(
          alo, ahi, 0, 1, 2, 3, 4, 5, 6, 7, 8, 9, 10, 11, 12, 13, 14, 15);
#pragma unroll
      for (int j = 0; j < 8; ++j) {
        v16h bb;
#pragma unroll
        for (int s = 0; s < 8; ++s) {
          bb[s]     = Bs[lhalf * 8 + s][j * 16 + lm];
          bb[s + 8] = Bs[16 + lhalf * 8 + s][j * 16 + lm];
        }
        acc[j] = __builtin_amdgcn_wmma_f32_16x16x32_f16(
            false, a, false, bb, (short)0, acc[j], false, false);
      }
    }

    // ---- bias (before LN, as in reference) ----
    const int mbase = wave * 16 + lhalf * 8;
    float bias_r[8];
#pragma unroll
    for (int r = 0; r < 8; ++r) bias_r[r] = bias[mbase + r];
#pragma unroll
    for (int j = 0; j < 8; ++j)
#pragma unroll
      for (int r = 0; r < 8; ++r) acc[j][r] += bias_r[r];

    // ---- LayerNorm over the 128 channels (block-wide via LDS atomics) ----
    __syncthreads();
    if (t < NTILE) { csum[t] = 0.f; csq[t] = 0.f; }
    __syncthreads();
#pragma unroll
    for (int j = 0; j < 8; ++j) {
      float s = 0.f, q = 0.f;
#pragma unroll
      for (int r = 0; r < 8; ++r) { const float v = acc[j][r]; s += v; q += v * v; }
      atomicAdd(&csum[j * 16 + lm], s);
      atomicAdd(&csq[j * 16 + lm], q);
    }
    __syncthreads();
    float gam_r[8], bet_r[8];
#pragma unroll
    for (int r = 0; r < 8; ++r) { gam_r[r] = gam[mbase + r]; bet_r[r] = bet[mbase + r]; }
#pragma unroll
    for (int j = 0; j < 8; ++j) {
      const int n   = j * 16 + lm;
      const float mean = csum[n] * (1.f / 128.f);
      const float var  = csq[n] * (1.f / 128.f) - mean * mean;
      const float rinv = rsqrtf(var + 1e-6f);
#pragma unroll
      for (int r = 0; r < 8; ++r) {
        const float val = (acc[j][r] - mean) * rinv;
        oacc[j][r] += g * (gam_r[r] * val + bet_r[r]);
      }
    }
  }

  // ---- store [B][C2][32][32], coalesced across lanes in N ----
#pragma unroll
  for (int j = 0; j < 8; ++j) {
    const int n = n0 + j * 16 + lm;
#pragma unroll
    for (int r = 0; r < 8; ++r) {
      const int m = wave * 16 + lhalf * 8 + r;
      out[((size_t)b * C2_ + m) * 1024 + n] = oacc[j][r];
    }
  }
}

// ---------------------------------------------------------------------------
extern "C" void kernel_launch(void* const* d_in, const int* in_sizes, int n_in,
                              void* d_out, int out_size, void* d_ws, size_t ws_size,
                              hipStream_t stream) {
  (void)in_sizes; (void)n_in; (void)out_size; (void)ws_size;
  const float* x        = (const float*)d_in[0];
  const float* w_gate   = (const float*)d_in[1];
  const float* expert_w = (const float*)d_in[2];
  const float* expert_b = (const float*)d_in[3];
  const float* eln_w    = (const float*)d_in[4];
  const float* eln_b    = (const float*)d_in[5];
  const float* shared_w = (const float*)d_in[6];
  const float* shared_b = (const float*)d_in[7];
  const float* sln_w    = (const float*)d_in[8];
  const float* sln_b    = (const float*)d_in[9];
  float* out = (float*)d_out;

  // workspace layout: gates[128] | x_f16 (16 MB) | w_f16 (720 KB)
  float*    gates = (float*)d_ws;
  _Float16* xh    = (_Float16*)((char*)d_ws + 1024);
  _Float16* wh    = xh + (size_t)B_ * C1_ * HW_ * HW_;

  moe_gate_kernel<<<B_, 256, 0, stream>>>(x, w_gate, gates);
  moe_convert_kernel<<<2048, 256, 0, stream>>>(x, expert_w, shared_w, xh, wh);
  moe_conv_wmma_kernel<<<B_ * 8, 256, 0, stream>>>(
      xh, wh, gates, expert_b, eln_w, eln_b, shared_b, sln_w, sln_b, out);
}
